// HSSM_linear_new_54116587930359
// MI455X (gfx1250) — matrix operand order repaired and checked by reading, not compiled, not run on hardware
//
#include <hip/hip_runtime.h>
#include <hip/hip_bf16.h>

// ---------------------------------------------------------------------------
// PVT-style pooled attention block for MI455X (gfx1250, wave32, WMMA).
// See analysis: bandwidth-bound at ~390MB; proj GEMM done in bf16 WMMA.
// ---------------------------------------------------------------------------

typedef __attribute__((ext_vector_type(16))) __bf16 v16bf;
typedef __attribute__((ext_vector_type(8)))  __bf16 v8bf;
typedef __attribute__((ext_vector_type(8)))  float  v8f;

#define BATCH 32
#define CDIM 320
#define HDIM 56
#define NTOK 3136          // 56*56
#define NHEAD 5
#define HEADD 64
#define MTOK 49            // 7*7 pooled
#define SCALE 0.125f       // 64^-0.5

// ---------------- Kernel A: per-token LN + q-projection + 8x8 mean pool ----
// grid: B*49 blocks (one per pooled cell), 256 threads.
// LDS: yt[320][65] f32 (pad 65 to break stride-64 bank conflicts),
//      qw[5][320], gg[320], bb[320].
#define A_LDS_BYTES (320*65*4 + 5*320*4 + 320*4 + 320*4)   // 92160

__global__ __launch_bounds__(256)
void ln_pool_q_kernel(const float* __restrict__ x,
                      const float* __restrict__ q_w,
                      const float* __restrict__ npre_g,
                      const float* __restrict__ npre_b,
                      float* __restrict__ q_ws,
                      float* __restrict__ pooled_ws)
{
    extern __shared__ char smem[];
    float* yt = (float*)smem;                   // [320][65]
    float* qw = (float*)(smem + 83200);         // [5][320]
    float* gg = (float*)(smem + 89600);         // [320]
    float* bb = (float*)(smem + 90880);         // [320]

    const int tid  = threadIdx.x;
    const int b    = blockIdx.x / 49;
    const int cell = blockIdx.x % 49;
    const int h0   = (cell / 7) * 8;
    const int w0   = (cell % 7) * 8;

    for (int i = tid; i < 5*320; i += 256) qw[i] = q_w[i];
    for (int i = tid; i < 320;   i += 256) { gg[i] = npre_g[i]; bb[i] = npre_b[i]; }

    // load x tile [c=320][8x8] (channel-major global layout)
    const float* xb = x + (size_t)b * CDIM * NTOK;
    for (int i = tid; i < 320*64; i += 256) {
        int c = i >> 6, s = i & 63;
        int hh = s >> 3, ww = s & 7;
        yt[c*65 + s] = xb[(size_t)c*NTOK + (h0+hh)*HDIM + (w0+ww)];
    }
    __syncthreads();

    // LN per token: 4 lanes per token (quad = lanes {4t..4t+3} within wave)
    const int t = tid >> 2, p = tid & 3;
    float sm = 0.f, sq = 0.f;
    #pragma unroll 8
    for (int i = 0; i < 80; ++i) {
        float v = yt[(p + 4*i)*65 + t];
        sm += v; sq += v*v;
    }
    sm += __shfl_xor(sm, 1, 32); sm += __shfl_xor(sm, 2, 32);
    sq += __shfl_xor(sq, 1, 32); sq += __shfl_xor(sq, 2, 32);
    const float mean = sm * (1.f/320.f);
    const float var  = sq * (1.f/320.f) - mean*mean;
    const float rstd = rsqrtf(var + 1e-5f);

    float qa[NHEAD] = {0.f,0.f,0.f,0.f,0.f};
    #pragma unroll 4
    for (int i = 0; i < 80; ++i) {
        int c = p + 4*i;
        float v = yt[c*65 + t];
        float y = (v - mean)*rstd*gg[c] + bb[c];
        yt[c*65 + t] = y;                        // own cells only: no race
        #pragma unroll
        for (int h = 0; h < NHEAD; ++h) qa[h] += y * qw[h*320 + c];
    }
    #pragma unroll
    for (int h = 0; h < NHEAD; ++h) {
        qa[h] += __shfl_xor(qa[h], 1, 32);
        qa[h] += __shfl_xor(qa[h], 2, 32);
    }
    if (p == 0) {
        int n = (h0 + (t >> 3))*HDIM + w0 + (t & 7);
        #pragma unroll
        for (int h = 0; h < NHEAD; ++h)
            q_ws[((size_t)b*NTOK + n)*NHEAD + h] = qa[h];
    }
    __syncthreads();

    // 8x8 mean pool of normalized y
    for (int c = tid; c < 320; c += 256) {
        float s = 0.f;
        #pragma unroll 8
        for (int s2 = 0; s2 < 64; ++s2) s += yt[c*65 + s2];
        pooled_ws[((size_t)b*MTOK + cell)*CDIM + c] = s * (1.f/64.f);
    }
}

// ---------------- Kernel B: sr(1x1 conv) + LN + GELU + k,v (+k min/max) ----
// grid: B blocks, 256 threads. LDS: pooled[49][320], xs[49][320], ks2[5][49]
#define B_LDS_BYTES (49*320*4 + 49*320*4 + 5*49*4 + 64)    // ~126.5KB

__global__ __launch_bounds__(256)
void sr_norm_kv_kernel(const float* __restrict__ pooled_ws,
                       const float* __restrict__ sr_w,
                       const float* __restrict__ sr_b,
                       const float* __restrict__ norm_g,
                       const float* __restrict__ norm_b,
                       const float* __restrict__ k_w,
                       const float* __restrict__ v_w,
                       float* __restrict__ kk_ws,
                       float* __restrict__ kmm_ws,
                       __bf16* __restrict__ vbf_ws)
{
    extern __shared__ char smem[];
    float* pl  = (float*)smem;                  // [49][320]
    float* xs  = (float*)(smem + 62720);        // [49][320]
    float* ks2 = (float*)(smem + 125440);       // [5][49]

    const int tid = threadIdx.x;
    const int b   = blockIdx.x;

    for (int i = tid; i < MTOK*CDIM; i += 256)
        pl[i] = pooled_ws[(size_t)b*MTOK*CDIM + i];
    __syncthreads();

    // x_ = pooled @ sr_w.T + sr_b   (49x320 outputs, K=320; tiny)
    for (int i = tid; i < MTOK*CDIM; i += 256) {
        int m = i / CDIM, c = i % CDIM;
        float acc = sr_b[c];
        const float* wr = sr_w + (size_t)c*CDIM;
        const float* pr = pl + m*CDIM;
        #pragma unroll 4
        for (int ci = 0; ci < CDIM; ++ci) acc += pr[ci]*wr[ci];
        xs[i] = acc;
    }
    __syncthreads();

    // LN + exact GELU, one wave per pooled token
    const int wv = tid >> 5, lane = tid & 31;
    for (int t = wv; t < MTOK; t += 8) {
        float sm = 0.f, sq = 0.f;
        #pragma unroll
        for (int i = 0; i < 10; ++i) {
            float v = xs[t*CDIM + lane + 32*i];
            sm += v; sq += v*v;
        }
        #pragma unroll
        for (int o = 1; o < 32; o <<= 1) {
            sm += __shfl_xor(sm, o, 32);
            sq += __shfl_xor(sq, o, 32);
        }
        float mean = sm*(1.f/320.f);
        float var  = sq*(1.f/320.f) - mean*mean;
        float rstd = rsqrtf(var + 1e-5f);
        #pragma unroll
        for (int i = 0; i < 10; ++i) {
            int c = lane + 32*i;
            float v = (xs[t*CDIM + c] - mean)*rstd*norm_g[c] + norm_b[c];
            float g = 0.5f*v*(1.0f + erff(v*0.70710678118f));   // exact GELU
            xs[t*CDIM + c] = g;
        }
    }
    __syncthreads();

    // k = x_ @ k_w.T  [5,49]
    for (int i = tid; i < NHEAD*MTOK; i += 256) {
        int h = i / MTOK, m = i % MTOK;
        float acc = 0.f;
        const float* wr = k_w + h*CDIM;
        const float* xr = xs + m*CDIM;
        #pragma unroll 4
        for (int c = 0; c < CDIM; ++c) acc += xr[c]*wr[c];
        ks2[i] = acc;
        kk_ws[(size_t)b*NHEAD*MTOK + i] = acc;
    }
    // v = x_ @ v_w.T  [49,320] -> bf16
    for (int i = tid; i < MTOK*CDIM; i += 256) {
        int m = i / CDIM, c = i % CDIM;
        float acc = 0.f;
        const float* wr = v_w + (size_t)c*CDIM;
        const float* xr = xs + m*CDIM;
        #pragma unroll 4
        for (int ci = 0; ci < CDIM; ++ci) acc += xr[ci]*wr[ci];
        vbf_ws[(size_t)b*MTOK*CDIM + i] = (__bf16)acc;
    }
    __syncthreads();
    if (tid < NHEAD) {
        float mn = 1e30f, mx = -1e30f;
        for (int m = 0; m < MTOK; ++m) {
            float v = ks2[tid*MTOK + m];
            mn = fminf(mn, v); mx = fmaxf(mx, v);
        }
        kmm_ws[b*2*NHEAD + tid*2 + 0] = mn;
        kmm_ws[b*2*NHEAD + tid*2 + 1] = mx;
    }
}

// ---------------- Kernel D: proj_w fp32 -> bf16 ---------------------------
__global__ __launch_bounds__(256)
void cvt_bf16_kernel(const float* __restrict__ src, __bf16* __restrict__ dst, int n)
{
    int i = blockIdx.x*blockDim.x + threadIdx.x;
    if (i < n) dst[i] = (__bf16)src[i];
}

// ---------------- Kernel C: rank-1 softmax + attn*v + WMMA proj + residual -
// grid: B*49 blocks (64 consecutive tokens each), 256 threads = 8 waves.
// LDS layout (bytes):
//   at   bf16 [64][344]  @ 0       (attnout A-tile, 16B-aligned rows)
//   vb   bf16 [49][328]  @ 44032
//   bt   bf16 [320][40]  @ 76176   (B chunk: proj_w rows x 32 c_in)
//   qs   f32  [64][5]    @ 101776
//   ks   f32  [5][49]    @ 103056
//   kmm  f32  [10]       @ 104048
//   osd  f32  [64][324]  @ 0       (overlay after WMMA K-loop)
#define AT_PITCH 344
#define VB_PITCH 328
#define BT_PITCH 40
#define OS_PITCH 324
#define C_LDS_BYTES 104448

__global__ __launch_bounds__(256)
void attn_proj_kernel(const float* __restrict__ x,
                      const float* __restrict__ q_ws,
                      const float* __restrict__ kk_ws,
                      const float* __restrict__ kmm_ws,
                      const __bf16* __restrict__ vbf_ws,
                      const __bf16* __restrict__ pwbf,
                      const float* __restrict__ proj_b,
                      float* __restrict__ out)
{
    extern __shared__ char smem[];
    __bf16* at  = (__bf16*)(smem + 0);
    __bf16* vb  = (__bf16*)(smem + 44032);
    __bf16* bt  = (__bf16*)(smem + 76176);
    float*  qs  = (float*)(smem + 101776);
    float*  ks  = (float*)(smem + 103056);
    float*  kmm = (float*)(smem + 104048);
    float*  osd = (float*)(smem);              // overlay, used after K-loop

    const int tid  = threadIdx.x;
    const int b    = blockIdx.x / 49;
    const int tile = blockIdx.x % 49;
    const int n0   = tile * 64;

    // ---- stage q, k, kminmax, v ----
    for (int i = tid; i < 64*NHEAD; i += 256)
        qs[i] = q_ws[((size_t)b*NTOK + n0 + (i/NHEAD))*NHEAD + (i%NHEAD)];
    for (int i = tid; i < NHEAD*MTOK; i += 256)
        ks[i] = kk_ws[(size_t)b*NHEAD*MTOK + i];
    if (tid < 2*NHEAD) kmm[tid] = kmm_ws[b*2*NHEAD + tid];
    {
        const uint32_t* src = (const uint32_t*)(vbf_ws + (size_t)b*MTOK*CDIM);
        for (int i = tid; i < MTOK*(CDIM/2); i += 256) {
            int m = i / (CDIM/2), j = i % (CDIM/2);
            ((uint32_t*)vb)[m*(VB_PITCH/2) + j] = src[i];
        }
    }
    __syncthreads();

    // ---- phase 1: rank-1 softmax + attn*v into bf16 A-tile ----
    {
        const int t = tid >> 2;      // token in tile
        const int p = tid & 3;       // quarter of head-dim
        float pm[MTOK];
        #pragma unroll
        for (int h = 0; h < NHEAD; ++h) {
            float qv = qs[t*NHEAD + h] * SCALE;
            float mx = (qv >= 0.f) ? qv*kmm[h*2+1] : qv*kmm[h*2+0];
            float Z = 0.f;
            for (int m = 0; m < MTOK; ++m) {
                float e = __expf(qv * ks[h*MTOK + m] - mx);
                pm[m] = e; Z += e;
            }
            float iz = 1.0f / Z;
            for (int j = 0; j < 16; ++j) {
                int c = h*HEADD + p*16 + j;
                float acc = 0.f;
                for (int m = 0; m < MTOK; ++m)
                    acc += pm[m] * (float)vb[m*VB_PITCH + c];
                at[t*AT_PITCH + c] = (__bf16)(acc * iz);
            }
        }
    }

    // ---- phase 2: WMMA proj GEMM: [64x320] x [320x320] ----
    const int lane = tid & 31;
    const int wv   = tid >> 5;
    const int tm   = wv & 3;            // token subtile (16 rows)
    const int ch   = wv >> 2;           // c_out half (10 N-tiles each)

    v8f acc[10] = {};

    // CDNA5 16-bit A(16x32) fragment: lane L -> row M=L&15, kbase=(L>>4)*8;
    // halves 0..7 = K kbase..kbase+7, halves 8..15 = K kbase+16..kbase+23.
    const int arow = tm*16 + (lane & 15);
    const int kb   = (lane >> 4) * 8;
    // CDNA5 16-bit B(32x16) fragment: lane L -> col N=L&15, same K order as A.
    const int bn   = lane & 15;
    const int kb2  = (lane >> 4) * 8;

    for (int kc = 0; kc < 10; ++kc) {
        __syncthreads();   // protect bt (and first time: phase-1 at writes)
        // stage B chunk: proj_w rows [0..319], c_in [kc*32 .. +31] (bf16)
        {
            const uint32_t* src = (const uint32_t*)pwbf;
            for (int i = tid; i < 320*16; i += 256) {
                int r = i >> 4, jc = i & 15;
                ((uint32_t*)bt)[r*(BT_PITCH/2) + jc] = src[r*(CDIM/2) + kc*16 + jc];
            }
        }
        __syncthreads();

        v8bf alo = *(const v8bf*)(at + arow*AT_PITCH + kc*32 + kb);
        v8bf ahi = *(const v8bf*)(at + arow*AT_PITCH + kc*32 + kb + 16);
        v16bf afrag = __builtin_shufflevector(alo, ahi,
            0,1,2,3,4,5,6,7,8,9,10,11,12,13,14,15);

        #pragma unroll
        for (int j = 0; j < 10; ++j) {
            int nt = ch*10 + j;
            v8bf blo = *(const v8bf*)(bt + (nt*16 + bn)*BT_PITCH + kb2);
            v8bf bhi = *(const v8bf*)(bt + (nt*16 + bn)*BT_PITCH + kb2 + 16);
            v16bf bfrag = __builtin_shufflevector(blo, bhi,
                0,1,2,3,4,5,6,7,8,9,10,11,12,13,14,15);
            acc[j] = __builtin_amdgcn_wmma_f32_16x16x32_bf16(
                false, afrag, false, bfrag, (short)0, acc[j], false, false);
        }
    }
    __syncthreads();   // at/vb/bt dead; overlay as f32 out-stage

    // C/D f32 16x16 layout: VGPR r, lane L -> M = r + (L>>4)*8, N = L&15
    {
        const int mbase = (lane >> 4) * 8;
        const int col0  = lane & 15;
        #pragma unroll
        for (int j = 0; j < 10; ++j) {
            int nt = ch*10 + j;
            #pragma unroll
            for (int r = 0; r < 8; ++r) {
                int trow = tm*16 + mbase + r;
                osd[trow*OS_PITCH + nt*16 + col0] = acc[j][r];
            }
        }
    }
    __syncthreads();

    // fused bias + residual, coalesced along tokens (output is [B,C,H,W])
    for (int i = tid; i < CDIM*64; i += 256) {
        int c = i >> 6, t = i & 63;
        size_t gi = ((size_t)b*CDIM + c)*NTOK + n0 + t;
        out[gi] = x[gi] + osd[t*OS_PITCH + c] + proj_b[c];
    }
}

// ---------------------------------------------------------------------------
// workspace offsets (bytes)
#define O_Q     0u                    // B*N*5 f32        = 2,007,040
#define O_POOL  2007040u              // B*49*320 f32     = 2,007,040
#define O_KK    4014080u              // B*5*49 f32       =    31,360
#define O_KMM   4045440u              // B*10 f32         =     1,280
#define O_VBF   4046848u              // B*49*320 bf16    = 1,003,520
#define O_PWBF  5050368u              // 320*320 bf16     =   204,800
// total ~5.26 MB

extern "C" void kernel_launch(void* const* d_in, const int* in_sizes, int n_in,
                              void* d_out, int out_size, void* d_ws, size_t ws_size,
                              hipStream_t stream) {
    const float* x      = (const float*)d_in[0];
    const float* q_w    = (const float*)d_in[1];
    const float* k_w    = (const float*)d_in[2];
    const float* v_w    = (const float*)d_in[3];
    const float* proj_w = (const float*)d_in[4];
    const float* proj_b = (const float*)d_in[5];
    const float* sr_w   = (const float*)d_in[6];
    const float* sr_b   = (const float*)d_in[7];
    const float* norm_g = (const float*)d_in[8];
    const float* norm_b = (const float*)d_in[9];
    const float* npre_g = (const float*)d_in[10];
    const float* npre_b = (const float*)d_in[11];
    float* out = (float*)d_out;

    char* ws = (char*)d_ws;
    float*  q_ws      = (float*)(ws + O_Q);
    float*  pooled_ws = (float*)(ws + O_POOL);
    float*  kk_ws     = (float*)(ws + O_KK);
    float*  kmm_ws    = (float*)(ws + O_KMM);
    __bf16* vbf_ws    = (__bf16*)(ws + O_VBF);
    __bf16* pwbf_ws   = (__bf16*)(ws + O_PWBF);

    // proj_w -> bf16 (L2-resident thereafter)
    cvt_bf16_kernel<<<(CDIM*CDIM + 255)/256, 256, 0, stream>>>(proj_w, pwbf_ws, CDIM*CDIM);

    // LN + q + pool
    ln_pool_q_kernel<<<BATCH*49, 256, A_LDS_BYTES, stream>>>(
        x, q_w, npre_g, npre_b, q_ws, pooled_ws);

    // sr conv + LN + GELU + k,v
    sr_norm_kv_kernel<<<BATCH, 256, B_LDS_BYTES, stream>>>(
        pooled_ws, sr_w, sr_b, norm_g, norm_b, k_w, v_w, kk_ws, kmm_ws, vbf_ws);

    // fused softmax + attn*v + WMMA proj + residual
    attn_proj_kernel<<<BATCH*49, 256, C_LDS_BYTES, stream>>>(
        x, q_ws, kk_ws, kmm_ws, vbf_ws, pwbf_ws, proj_b, out);
}
